// XPUFluxTransformer2dModel_53884659696164
// MI455X (gfx1250) — compile-verified, matrix operand order checked by reading, not run on hardware
//
#include <hip/hip_runtime.h>
#include <hip/hip_bf16.h>

// ---------------------------------------------------------------------------
// Types for CDNA5 WMMA (wave32): v_wmma_f32_16x16x32_f16
// ---------------------------------------------------------------------------
typedef __attribute__((ext_vector_type(16))) _Float16 v16h;
typedef __attribute__((ext_vector_type(8)))  _Float16 v8h;
typedef __attribute__((ext_vector_type(4)))  _Float16 v4h;
typedef __attribute__((ext_vector_type(2)))  _Float16 v2h;
typedef __attribute__((ext_vector_type(2)))  __fp16   v2fp;
typedef __attribute__((ext_vector_type(8)))  float    v8f;
typedef __attribute__((ext_vector_type(4)))  unsigned int v4u;
typedef __attribute__((ext_vector_type(4)))  int      v4i;
typedef __attribute__((ext_vector_type(8)))  int      v8i;

static __device__ __forceinline__ v16h cat16(v8h lo, v8h hi) {
  return __builtin_shufflevector(lo, hi, 0,1,2,3,4,5,6,7,8,9,10,11,12,13,14,15);
}
static __device__ __forceinline__ v2h pk2(float a, float b) {
  v2fp r = __builtin_amdgcn_cvt_pkrtz(a, b);   // v_cvt_pk_f16_f32 (RTZ)
  return __builtin_bit_cast(v2h, r);
}
static __device__ __forceinline__ v8f zero8() {
  v8f v;
  #pragma unroll
  for (int i = 0; i < 8; ++i) v[i] = 0.f;
  return v;
}
static __device__ __forceinline__ float gelu_tanh(float x) {
  float x3 = x * x * x;
  return 0.5f * x * (1.f + tanhf(0.7978845608028654f * (x + 0.044715f * x3)));
}

#define H_DIM   3072
#define MLP_DIM 12288
#define L_TXT   512
#define L_IMG   1024
#define L_ALL   1536
#define NHEADS  24
#define HD      128

// ---------------------------------------------------------------------------
// Tensor Data Mover: 2D tile load Global -> LDS (f16 elements, data_size=2B).
// Descriptor per CDNA5 ISA 8.3-8.5. Issued once per wave; EXEC ignored.
//   lds_off     : LDS byte address of tile destination (packed rows)
//   gaddr       : global byte address of tile start
//   tensor_d0/1 : tensor extent (elements) in dim0 / dim1 (for OOB clamping)
//   tile_d0/1   : tile extent (elements); tile rows packed into LDS
//   stride0     : dim0 stride (elements) between consecutive dim1 lines
// ---------------------------------------------------------------------------
static __device__ __forceinline__ void tdm_load_2d_f16(
    unsigned lds_off, const void* gaddr,
    unsigned tensor_d0, unsigned tensor_d1,
    unsigned tile_d0, unsigned tile_d1, unsigned stride0) {
  unsigned long long ga = (unsigned long long)(uintptr_t)gaddr;
  v4u g0;
  g0[0] = 1u;                                        // count=1 (valid user D#)
  g0[1] = lds_off;                                   // lds_addr [63:32]
  g0[2] = (unsigned)(ga & 0xFFFFFFFFu);              // global_addr lo
  g0[3] = (unsigned)((ga >> 32) & 0x1FFFFFFu)        // global_addr [56:32]
        | (2u << 30);                                // type=2 ("image")
  v8i g1;
  g1[0] = (int)(1u << 16);                           // data_size=1 (2 bytes)
  g1[1] = (int)((tensor_d0 & 0xFFFFu) << 16);        // tensor_dim0[15:0] @ 63:48
  g1[2] = (int)(((tensor_d0 >> 16) & 0xFFFFu)        // tensor_dim0[31:16]
        | ((tensor_d1 & 0xFFFFu) << 16));            // tensor_dim1[15:0]
  g1[3] = (int)(((tensor_d1 >> 16) & 0xFFFFu)        // tensor_dim1[31:16]
        | ((tile_d0 & 0xFFFFu) << 16));              // tile_dim0 @ 127:112
  g1[4] = (int)(tile_d1 & 0xFFFFu);                  // tile_dim1; tile_dim2=0
  g1[5] = (int)stride0;                              // tensor_dim0_stride[31:0]
  g1[6] = 0;                                         // stride hi, dim1_stride lo
  g1[7] = 0;
  v4i z4 = {0, 0, 0, 0};
#if defined(__clang_major__) && (__clang_major__ >= 23)
  v8i z8 = {0, 0, 0, 0, 0, 0, 0, 0};
  __builtin_amdgcn_tensor_load_to_lds(g0, g1, z4, z4, z8, 0);
#else
  __builtin_amdgcn_tensor_load_to_lds(g0, g1, z4, z4, 0);
#endif
}

// ---------------------------------------------------------------------------
// mod kernel: out[col] = bias[col] + sum_k silu(vec[k]) * W[k][col]
// ---------------------------------------------------------------------------
__global__ __launch_bounds__(128) void mod_kernel(
    const float* __restrict__ vec, const float* __restrict__ W,
    const float* __restrict__ b, float* __restrict__ out) {
  __shared__ float sv[H_DIM];
  int tid = threadIdx.x;
  for (int i = tid; i < H_DIM; i += 128) {
    float x = vec[i];
    sv[i] = x / (1.f + __expf(-x));
  }
  __syncthreads();
  int col = blockIdx.x * 128 + tid;
  float acc = b[col];
  for (int k = 0; k < H_DIM; ++k)
    acc = fmaf(sv[k], W[(size_t)k * (6 * H_DIM) + col], acc);
  out[col] = acc;
}

// ---------------------------------------------------------------------------
// LayerNorm + modulate: Y[row] = LN(X[row]) * (1 + sc) + sh
// ---------------------------------------------------------------------------
__global__ __launch_bounds__(256) void lnmod_kernel(
    const float* __restrict__ X, const float* __restrict__ sh,
    const float* __restrict__ sc, float* __restrict__ Y) {
  __shared__ float red[256];
  __shared__ float s_mean, s_rstd;
  int tid = threadIdx.x;
  const float* x = X + (size_t)blockIdx.x * H_DIM;
  float s = 0.f;
  for (int i = tid; i < H_DIM; i += 256) s += x[i];
  red[tid] = s; __syncthreads();
  for (int w = 128; w > 0; w >>= 1) { if (tid < w) red[tid] += red[tid + w]; __syncthreads(); }
  if (tid == 0) s_mean = red[0] * (1.f / H_DIM);
  __syncthreads();
  float mean = s_mean;
  float v = 0.f;
  for (int i = tid; i < H_DIM; i += 256) { float d = x[i] - mean; v += d * d; }
  red[tid] = v; __syncthreads();
  for (int w = 128; w > 0; w >>= 1) { if (tid < w) red[tid] += red[tid + w]; __syncthreads(); }
  if (tid == 0) s_rstd = rsqrtf(red[0] * (1.f / H_DIM) + 1e-6f);
  __syncthreads();
  float rstd = s_rstd;
  float* y = Y + (size_t)blockIdx.x * H_DIM;
  for (int i = tid; i < H_DIM; i += 256)
    y[i] = (x[i] - mean) * rstd * (1.f + sc[i]) + sh[i];
}

// ---------------------------------------------------------------------------
// Tiled WMMA GEMM: C[M,N] = epi(A[M,K] @ B[K,N] + bias)
//   EPI 0: acc + bias | EPI 1: gelu(acc + bias) | EPI 2: res + gate*(acc+bias)
// Block tile 128x64x32, 256 threads = 8 waves, each wave 32x32 via 2x2 WMMA.
// fp32 -> f16 staging uses v_cvt_pk packed converts + wide DS stores.
// ---------------------------------------------------------------------------
#define BM 128
#define BN 64
#define BK 32

template <int EPI>
__global__ __launch_bounds__(256) void gemm_wmma(
    const float* __restrict__ A, const float* __restrict__ B,
    const float* __restrict__ bias, const float* __restrict__ res,
    const float* __restrict__ gate, float* __restrict__ C,
    int M, int N, int K) {
  __shared__ _Float16 As[BM * BK];
  __shared__ _Float16 Bs[BN * BK];
  int tid  = threadIdx.x;
  int lane = tid & 31;
  int wave = tid >> 5;
  int wm   = wave & 3;        // wave m-tile: +32*wm rows
  int wn   = wave >> 2;       // wave n-tile: +32*wn cols
  int lg   = lane >> 4;       // lane group (0/1)
  int lr   = lane & 15;
  int bm   = blockIdx.x * BM;
  int bn   = blockIdx.y * BN;

  v8f acc[2][2];
  #pragma unroll
  for (int i = 0; i < 2; ++i)
    #pragma unroll
    for (int j = 0; j < 2; ++j) acc[i][j] = zero8();

  for (int k0 = 0; k0 < K; k0 += BK) {
    // --- stage A tile (128x32 fp32 -> f16): packed cvt, ds_store_b64 ---
    #pragma unroll
    for (int i = 0; i < 4; ++i) {
      int c  = tid + i * 256;       // 0..1023 float4 chunks
      int r  = c >> 3;
      int kc = (c & 7) << 2;
      const float4 f = *(const float4*)(A + (size_t)(bm + r) * K + k0 + kc);
      v2h p0 = pk2(f.x, f.y);
      v2h p1 = pk2(f.z, f.w);
      *(v4h*)&As[r * BK + kc] = __builtin_shufflevector(p0, p1, 0, 1, 2, 3);
    }
    // --- stage B tile (32x64 fp32 -> f16 transposed to Bs[n][k]) ---
    // Each thread: one k-pair x 4 cols -> 4 packed ds_store_b32.
    {
      int kk = (tid >> 4) << 1;     // 0,2,..,30
      int nn = (tid & 15) << 2;     // 0,4,..,60
      const float4 f0 = *(const float4*)(B + (size_t)(k0 + kk) * N + bn + nn);
      const float4 f1 = *(const float4*)(B + (size_t)(k0 + kk + 1) * N + bn + nn);
      *(v2h*)&Bs[(nn + 0) * BK + kk] = pk2(f0.x, f1.x);
      *(v2h*)&Bs[(nn + 1) * BK + kk] = pk2(f0.y, f1.y);
      *(v2h*)&Bs[(nn + 2) * BK + kk] = pk2(f0.z, f1.z);
      *(v2h*)&Bs[(nn + 3) * BK + kk] = pk2(f0.w, f1.w);
    }
    // prefetch next K-slab of B (global_prefetch_b8)
    if (k0 + BK < K)
      __builtin_prefetch(B + (size_t)(k0 + BK + (tid >> 4)) * N + bn + (tid & 15) * 4, 0, 1);
    __syncthreads();

    v16h af[2], bf[2];
    #pragma unroll
    for (int mt = 0; mt < 2; ++mt) {
      const _Float16* ap = &As[(wm * 32 + mt * 16 + lr) * BK + lg * 8];
      af[mt] = cat16(*(const v8h*)ap, *(const v8h*)(ap + 16));
    }
    #pragma unroll
    for (int nt = 0; nt < 2; ++nt) {
      const _Float16* bp = &Bs[(wn * 32 + nt * 16 + lr) * BK + lg * 8];
      bf[nt] = cat16(*(const v8h*)bp, *(const v8h*)(bp + 16));
    }
    #pragma unroll
    for (int mt = 0; mt < 2; ++mt)
      #pragma unroll
      for (int nt = 0; nt < 2; ++nt)
        acc[mt][nt] = __builtin_amdgcn_wmma_f32_16x16x32_f16(
            false, af[mt], false, bf[nt], (short)0, acc[mt][nt], false, false);
    __syncthreads();
  }

  // --- epilogue: C layout per ISA 7.12.2 (VGPR r, lane l -> M=r+8*lg, N=lr)
  #pragma unroll
  for (int mt = 0; mt < 2; ++mt) {
    #pragma unroll
    for (int nt = 0; nt < 2; ++nt) {
      int col = bn + wn * 32 + nt * 16 + lr;
      float bv = bias[col];
      float gv = (EPI == 2) ? gate[col] : 0.f;
      #pragma unroll
      for (int r = 0; r < 8; ++r) {
        int row = bm + wm * 32 + mt * 16 + r + 8 * lg;
        float v = acc[mt][nt][r] + bv;
        if (EPI == 1) v = gelu_tanh(v);
        if (EPI == 2) v = res[(size_t)row * N + col] + gv * v;
        C[(size_t)row * N + col] = v;
      }
    }
  }
}

// ---------------------------------------------------------------------------
// RMS-norm + RoPE + repack for attention.
// Outputs f16: Qh[h][t][d], Kh[h][t][d], Vt[h][d][t].
// ---------------------------------------------------------------------------
__global__ __launch_bounds__(128) void rmsrope_kernel(
    const float* __restrict__ qkv_txt, const float* __restrict__ qkv_img,
    const float* __restrict__ nqw_t, const float* __restrict__ nkw_t,
    const float* __restrict__ nqw_i, const float* __restrict__ nkw_i,
    const float* __restrict__ pe,
    _Float16* __restrict__ Qh, _Float16* __restrict__ Kh, _Float16* __restrict__ Vt) {
  int t = blockIdx.x, h = blockIdx.y, d = threadIdx.x;
  const float *src, *qw, *kw;
  if (t < L_TXT) { src = qkv_txt + (size_t)t * (3 * H_DIM); qw = nqw_t; kw = nkw_t; }
  else           { src = qkv_img + (size_t)(t - L_TXT) * (3 * H_DIM); qw = nqw_i; kw = nkw_i; }
  float q = src[h * HD + d];
  float k = src[H_DIM + h * HD + d];
  float v = src[2 * H_DIM + h * HD + d];

  __shared__ float red[128];
  __shared__ float bc[2];
  __shared__ float sq[128], sk[128];
  red[d] = q * q; __syncthreads();
  for (int w = 64; w > 0; w >>= 1) { if (d < w) red[d] += red[d + w]; __syncthreads(); }
  if (d == 0) bc[0] = rsqrtf(red[0] * (1.f / HD) + 1e-6f);
  __syncthreads();
  red[d] = k * k; __syncthreads();
  for (int w = 64; w > 0; w >>= 1) { if (d < w) red[d] += red[d + w]; __syncthreads(); }
  if (d == 0) bc[1] = rsqrtf(red[0] * (1.f / HD) + 1e-6f);
  __syncthreads();
  sq[d] = q * bc[0] * qw[d];
  sk[d] = k * bc[1] * kw[d];
  __syncthreads();
  int i = d >> 1;
  const float* f = pe + ((size_t)t * 64 + i) * 4;
  float x0q = sq[2 * i], x1q = sq[2 * i + 1];
  float x0k = sk[2 * i], x1k = sk[2 * i + 1];
  float oq = (d & 1) ? (f[2] * x0q + f[3] * x1q) : (f[0] * x0q + f[1] * x1q);
  float ok = (d & 1) ? (f[2] * x0k + f[3] * x1k) : (f[0] * x0k + f[1] * x1k);
  size_t td = ((size_t)h * L_ALL + t) * HD + d;
  Qh[td] = (_Float16)oq;
  Kh[td] = (_Float16)ok;
  Vt[((size_t)h * HD + d) * L_ALL + t] = (_Float16)v;
}

// ---------------------------------------------------------------------------
// Flash attention with TDM-staged K/V tiles in LDS.
// grid (L/64, 24 heads), 128 threads (4 waves); each wave owns 16 query rows.
// Per 32-key block: wave0 DMAs K tile (32x128) and V^T tile (128x32) into LDS
// via TENSOR_LOAD_TO_LDS + s_wait_tensorcnt, barrier-publishes to all waves.
// QK^T and PV both in WMMA f16; online softmax with lane shuffles; P restaged
// through LDS into the A-fragment layout.
// ---------------------------------------------------------------------------
__global__ __launch_bounds__(128) void attn_kernel(
    const _Float16* __restrict__ Qh, const _Float16* __restrict__ Kh,
    const _Float16* __restrict__ Vt, float* __restrict__ Aout) {
  const float scale = 0.08838834764831845f;  // 1/sqrt(128)
  int h = blockIdx.y;
  int wave = threadIdx.x >> 5;
  int lane = threadIdx.x & 31;
  int lg = lane >> 4, lr = lane & 15;
  int q0 = blockIdx.x * 64 + wave * 16;
  const _Float16* Q = Qh + (size_t)h * L_ALL * HD;
  const _Float16* K = Kh + (size_t)h * L_ALL * HD;
  const _Float16* V = Vt + (size_t)h * HD * L_ALL;

  __shared__ _Float16 Ks[32 * HD];          // [key][d]  : 8 KB
  __shared__ _Float16 Vs[HD * 32];          // [d][key]  : 8 KB
  __shared__ _Float16 Pbuf[4][16 * 32];     //           : 4 KB
  _Float16* P = Pbuf[wave];
  unsigned ks_off = (unsigned)(uintptr_t)&Ks[0];   // LDS byte address
  unsigned vs_off = (unsigned)(uintptr_t)&Vs[0];

  // Q fragments: 4 d-slices of 32
  v16h qf[4];
  #pragma unroll
  for (int s = 0; s < 4; ++s) {
    const _Float16* qp = Q + (size_t)(q0 + lr) * HD + s * 32 + lg * 8;
    qf[s] = cat16(*(const v8h*)qp, *(const v8h*)(qp + 16));
  }

  float mstat[8], lstat[8];
  v8f o[8];
  #pragma unroll
  for (int r = 0; r < 8; ++r) { mstat[r] = -1e30f; lstat[r] = 0.f; o[r] = zero8(); }

  for (int kb = 0; kb < L_ALL; kb += 32) {
    // --- TDM: stage K tile (32 keys x 128 d) and V^T tile (128 d x 32 keys)
    if (wave == 0) {
      tdm_load_2d_f16(ks_off, K + (size_t)kb * HD, HD, L_ALL, HD, 32, HD);
      tdm_load_2d_f16(vs_off, V + kb, L_ALL, HD, 32, HD, L_ALL);
      __builtin_amdgcn_s_wait_tensorcnt(0);
    }
    __syncthreads();

    v8f s0 = zero8(), s1 = zero8();
    #pragma unroll
    for (int s = 0; s < 4; ++s) {
      const _Float16* kp0 = &Ks[lr * HD + s * 32 + lg * 8];
      v16h b0 = cat16(*(const v8h*)kp0, *(const v8h*)(kp0 + 16));
      s0 = __builtin_amdgcn_wmma_f32_16x16x32_f16(false, qf[s], false, b0, (short)0, s0, false, false);
      const _Float16* kp1 = &Ks[(16 + lr) * HD + s * 32 + lg * 8];
      v16h b1 = cat16(*(const v8h*)kp1, *(const v8h*)(kp1 + 16));
      s1 = __builtin_amdgcn_wmma_f32_16x16x32_f16(false, qf[s], false, b1, (short)0, s1, false, false);
    }
    // online softmax per row r (rows spread across 16 lanes of half-wave)
    #pragma unroll
    for (int r = 0; r < 8; ++r) {
      float a = s0[r] * scale, b = s1[r] * scale;
      float mx = fmaxf(a, b);
      #pragma unroll
      for (int off = 1; off < 16; off <<= 1) mx = fmaxf(mx, __shfl_xor(mx, off, 32));
      float mn = fmaxf(mstat[r], mx);
      float alpha = __expf(mstat[r] - mn);
      float pa = __expf(a - mn), pb = __expf(b - mn);
      float sm = pa + pb;
      #pragma unroll
      for (int off = 1; off < 16; off <<= 1) sm += __shfl_xor(sm, off, 32);
      lstat[r] = lstat[r] * alpha + sm;
      mstat[r] = mn;
      #pragma unroll
      for (int nt = 0; nt < 8; ++nt) o[nt][r] *= alpha;
      P[(r + 8 * lg) * 32 + lr]      = (_Float16)pa;
      P[(r + 8 * lg) * 32 + 16 + lr] = (_Float16)pb;
    }
    // wave-internal cross-lane LDS dependency (DS in-order per wave)
    asm volatile("s_wait_dscnt 0" ::: "memory");
    const _Float16* pp = P + lr * 32 + lg * 8;
    v16h pf = cat16(*(const v8h*)pp, *(const v8h*)(pp + 16));
    #pragma unroll
    for (int nt = 0; nt < 8; ++nt) {
      const _Float16* vp = &Vs[(nt * 16 + lr) * 32 + lg * 8];
      v16h vf = cat16(*(const v8h*)vp, *(const v8h*)(vp + 16));
      o[nt] = __builtin_amdgcn_wmma_f32_16x16x32_f16(false, pf, false, vf, (short)0, o[nt], false, false);
    }
    __syncthreads();   // all waves done with Ks/Vs before next DMA
  }
  // normalize and write: Aout[t][h*128 + d]
  #pragma unroll
  for (int r = 0; r < 8; ++r) {
    float inv = 1.f / lstat[r];
    int t = q0 + r + 8 * lg;
    #pragma unroll
    for (int nt = 0; nt < 8; ++nt)
      Aout[(size_t)t * H_DIM + h * HD + nt * 16 + lr] = o[nt][r] * inv;
  }
}

// ---------------------------------------------------------------------------
// Host launch
// ---------------------------------------------------------------------------
extern "C" void kernel_launch(void* const* d_in, const int* in_sizes, int n_in,
                              void* d_out, int out_size, void* d_ws, size_t ws_size,
                              hipStream_t stream) {
  (void)in_sizes; (void)n_in; (void)out_size; (void)ws_size;
  const float* img      = (const float*)d_in[0];
  const float* txt      = (const float*)d_in[1];
  const float* vec      = (const float*)d_in[2];
  const float* pe       = (const float*)d_in[3];
  const float* mod_w    = (const float*)d_in[4];
  const float* mod_b    = (const float*)d_in[5];
  const float* qkv_w    = (const float*)d_in[6];
  const float* qkv_b    = (const float*)d_in[7];
  const float* norm_q_w = (const float*)d_in[8];
  const float* norm_k_w = (const float*)d_in[9];
  const float* out_w    = (const float*)d_in[10];
  const float* out_b    = (const float*)d_in[11];
  const float* fc1_w    = (const float*)d_in[12];
  const float* fc1_b    = (const float*)d_in[13];
  const float* fc2_w    = (const float*)d_in[14];
  const float* fc2_b    = (const float*)d_in[15];
  const float* mod_c_w  = (const float*)d_in[16];
  const float* mod_c_b  = (const float*)d_in[17];
  const float* qkv_c_w  = (const float*)d_in[18];
  const float* qkv_c_b  = (const float*)d_in[19];
  const float* norm_aq_w= (const float*)d_in[20];
  const float* norm_ak_w= (const float*)d_in[21];
  const float* out_c_w  = (const float*)d_in[22];
  const float* out_c_b  = (const float*)d_in[23];
  const float* fc1_c_w  = (const float*)d_in[24];
  const float* fc1_c_b  = (const float*)d_in[25];
  const float* fc2_c_w  = (const float*)d_in[26];
  const float* fc2_c_b  = (const float*)d_in[27];
  float* out = (float*)d_out;

  // workspace carve-up (256B aligned)
  char* base = (char*)d_ws;
  size_t off = 0;
  auto carve = [&](size_t bytes) { char* p = base + off; off += (bytes + 255) & ~(size_t)255; return p; };
  float*    im        = (float*)carve(6 * H_DIM * 4);
  float*    tm        = (float*)carve(6 * H_DIM * 4);
  float*    xmod_img  = (float*)carve((size_t)L_IMG * H_DIM * 4);
  float*    xmod_txt  = (float*)carve((size_t)L_TXT * H_DIM * 4);
  float*    qkv_img   = (float*)carve((size_t)L_IMG * 3 * H_DIM * 4);
  float*    qkv_txt   = (float*)carve((size_t)L_TXT * 3 * H_DIM * 4);
  _Float16* Qh        = (_Float16*)carve((size_t)NHEADS * L_ALL * HD * 2);
  _Float16* Kh        = (_Float16*)carve((size_t)NHEADS * L_ALL * HD * 2);
  _Float16* Vt        = (_Float16*)carve((size_t)NHEADS * HD * L_ALL * 2);
  float*    attn      = (float*)carve((size_t)L_ALL * H_DIM * 4);
  float*    img2      = (float*)carve((size_t)L_IMG * H_DIM * 4);
  float*    txt2      = (float*)carve((size_t)L_TXT * H_DIM * 4);
  float*    xmod2_img = (float*)carve((size_t)L_IMG * H_DIM * 4);
  float*    xmod2_txt = (float*)carve((size_t)L_TXT * H_DIM * 4);
  float*    h_img     = (float*)carve((size_t)L_IMG * MLP_DIM * 4);
  float*    h_txt     = (float*)carve((size_t)L_TXT * MLP_DIM * 4);

  // 1) modulation vectors
  mod_kernel<<<144, 128, 0, stream>>>(vec, mod_w,   mod_b,   im);
  mod_kernel<<<144, 128, 0, stream>>>(vec, mod_c_w, mod_c_b, tm);

  // 2) LN + modulate (split offsets: sh1=0, sc1=H, g1=2H, sh2=3H, sc2=4H, g2=5H)
  lnmod_kernel<<<L_IMG, 256, 0, stream>>>(img, im + 0, im + H_DIM, xmod_img);
  lnmod_kernel<<<L_TXT, 256, 0, stream>>>(txt, tm + 0, tm + H_DIM, xmod_txt);

  // 3) fused QKV GEMMs
  gemm_wmma<0><<<dim3(L_IMG / BM, 3 * H_DIM / BN), 256, 0, stream>>>(
      xmod_img, qkv_w, qkv_b, nullptr, nullptr, qkv_img, L_IMG, 3 * H_DIM, H_DIM);
  gemm_wmma<0><<<dim3(L_TXT / BM, 3 * H_DIM / BN), 256, 0, stream>>>(
      xmod_txt, qkv_c_w, qkv_c_b, nullptr, nullptr, qkv_txt, L_TXT, 3 * H_DIM, H_DIM);

  // 4) RMS + RoPE + repack (txt tokens first, then img — matches reference)
  rmsrope_kernel<<<dim3(L_ALL, NHEADS), 128, 0, stream>>>(
      qkv_txt, qkv_img, norm_aq_w, norm_ak_w, norm_q_w, norm_k_w, pe, Qh, Kh, Vt);

  // 5) attention
  attn_kernel<<<dim3(L_ALL / 64, NHEADS), 128, 0, stream>>>(Qh, Kh, Vt, attn);

  // 6) output projection with gated residual
  gemm_wmma<2><<<dim3(L_IMG / BM, H_DIM / BN), 256, 0, stream>>>(
      attn + (size_t)L_TXT * H_DIM, out_w, out_b, img, im + 2 * H_DIM, img2,
      L_IMG, H_DIM, H_DIM);
  gemm_wmma<2><<<dim3(L_TXT / BM, H_DIM / BN), 256, 0, stream>>>(
      attn, out_c_w, out_c_b, txt, tm + 2 * H_DIM, txt2, L_TXT, H_DIM, H_DIM);

  // 7) LN + modulate (second)
  lnmod_kernel<<<L_IMG, 256, 0, stream>>>(img2, im + 3 * H_DIM, im + 4 * H_DIM, xmod2_img);
  lnmod_kernel<<<L_TXT, 256, 0, stream>>>(txt2, tm + 3 * H_DIM, tm + 4 * H_DIM, xmod2_txt);

  // 8) MLP fc1 (+GELU)
  gemm_wmma<1><<<dim3(L_IMG / BM, MLP_DIM / BN), 256, 0, stream>>>(
      xmod2_img, fc1_w, fc1_b, nullptr, nullptr, h_img, L_IMG, MLP_DIM, H_DIM);
  gemm_wmma<1><<<dim3(L_TXT / BM, MLP_DIM / BN), 256, 0, stream>>>(
      xmod2_txt, fc1_c_w, fc1_c_b, nullptr, nullptr, h_txt, L_TXT, MLP_DIM, H_DIM);

  // 9) MLP fc2 with gated residual -> final outputs (img first, then txt)
  gemm_wmma<2><<<dim3(L_IMG / BM, H_DIM / BN), 256, 0, stream>>>(
      h_img, fc2_w, fc2_b, img2, im + 5 * H_DIM, out, L_IMG, H_DIM, MLP_DIM);
  gemm_wmma<2><<<dim3(L_TXT / BM, H_DIM / BN), 256, 0, stream>>>(
      h_txt, fc2_c_w, fc2_c_b, txt2, tm + 5 * H_DIM, out + (size_t)L_IMG * H_DIM,
      L_TXT, H_DIM, MLP_DIM);
}
